// CrossAttention_27943057227846
// MI455X (gfx1250) — matrix operand order repaired and checked by reading, not compiled, not run on hardware
//
#include <hip/hip_runtime.h>

// ============================================================================
// CrossAttention for MI455X (gfx1250, wave32, WMMA).
//
//   * All GEMMs via v_wmma_f32_16x16x32_bf16 (f32 -> bf16 inputs, f32 accum).
//   * Score matrix S (8192x8192 bf16 = 128 MB) is MATERIALIZED: fits the
//     192 MB L2, cheaper than the +50% FLOP recompute flash-attention would
//     need at head dim 1024.
//   * V projected into transposed layout VT[H][N2] so every GEMM is "NT"
//     (out[m,n] = sum_k A[m,k]*B[n,k]) -> contiguous LDS fragment reads.
//   * bf16-input GEMMs: DOUBLE-BUFFERED LDS fed by GLOBAL_LOAD_ASYNC_TO_LDS_B128
//     (ASYNCcnt); s_wait_asynccnt 8 retires tile t while tile t+1 is in flight.
//   * f32-input GEMMs: software pipeline through VGPRs (loads for tile t+1
//     issue before the compute barrier of tile t).
//   * Epilogues for row-major outputs staged through LDS -> coalesced
//     global_store_b128.
//   * Register blocking: WG tile 128x128, BK=64, each wave 32x64 ->
//     16 WMMA per LDS stage.
//
// Workspace layout (needs ~176.1 MB):
//   [0,16M)    Q   bf16 [8192 x 1024]
//   [16M,32M)  K   bf16 [8192 x 1024]
//   [32M,48M)  VT  bf16 [1024 x 8192]
//   [48M,176M) S/P bf16 [8192 x 8192]   (exp'd in place)
//   [176M,+32K) rowsum f32 [8192]
// ============================================================================

typedef __attribute__((ext_vector_type(16))) __bf16        v16bf;
typedef __attribute__((ext_vector_type(8)))  float         v8f;
typedef __attribute__((ext_vector_type(4)))  unsigned int  u32x4;
typedef __attribute__((ext_vector_type(4)))  float         f32x4;

union FragBF { v16bf v; u32x4 u[2]; };

__device__ __forceinline__ unsigned short f32_to_bf16(float f) {
  unsigned int u = __float_as_uint(f);
  u += 0x7FFFu + ((u >> 16) & 1u);          // round-to-nearest-even
  return (unsigned short)(u >> 16);
}
__device__ __forceinline__ float bf16_to_f32(unsigned int bits16) {
  return __uint_as_float(bits16 << 16);
}

// Async global->LDS copy (CDNA5, tracked by ASYNCcnt).  VDST holds the LDS
// byte address (low 32 bits of the generic shared pointer, ISA 10.2 aperture
// mapping); VADDR is the 64-bit global address.
__device__ __forceinline__ void async_copy_b128(unsigned lds_off, const void* g) {
  asm volatile("global_load_async_to_lds_b128 %0, %1, off"
               :: "v"(lds_off), "v"(g) : "memory");
}
template <int N>
__device__ __forceinline__ void wait_asynccnt() {
  asm volatile("s_wait_asynccnt %0" :: "n"(N) : "memory");
}

// GEMM modes
#define MODE_PROJ    0   // A,B f32; out = bf16(A@B^T + bias), row-major
#define MODE_PROJ_T  1   // A,B f32; out = bf16(A@B^T + bias), TRANSPOSED store
#define MODE_SCORE   2   // A,B bf16; out = bf16(scale * A@B^T)
#define MODE_OUT     3   // A,B bf16; out = f32((A@B^T) / rowsum[m])

#define BM 128
#define BN 128
#define BKT 64
#define LDT 72    // LDS tile row pitch (bf16 elems): 144 B, multiple of 16 B
#define SP16 136  // staging pitch, bf16 epilogue (272 B, mult of 16 B)
#define SP32 132  // staging pitch, f32 epilogue  (528 B, mult of 16 B)
#define TILE_E (BM * LDT)   // ushorts per tile buffer

template <int MODE>
__global__ __launch_bounds__(256) void gemm_nt(
    const void* __restrict__ Aptr, const void* __restrict__ Bptr,
    const float* __restrict__ bias, const float* __restrict__ rowsum,
    void* __restrict__ Optr, int Kdim, int ldo, float scale)
{
  // MODE<=1: single-buffer (A,B) + bf16 staging.  MODE>=2: double-buffer
  // (A0,B0,A1,B1) = 72 KB; f32 staging (128x132x4 = 66 KB) reuses it.
  __shared__ __align__(16) unsigned short sm[(MODE <= 1) ? (2 * TILE_E)
                                                         : (4 * TILE_E)];

  const int tid  = threadIdx.x;
  const int lane = tid & 31;
  const int wave = tid >> 5;
  const int lr   = lane & 15;          // 0..15
  const int lh   = lane >> 4;          // 0 or 1
  const int mw   = (wave >> 1) * 32;   // wave row offset in 128x128 tile
  const int nw   = (wave & 1) * 64;    // wave col offset
  const size_t m0 = (size_t)blockIdx.y * BM;
  const size_t n0 = (size_t)blockIdx.x * BN;

  v8f acc[2][4];
#pragma unroll
  for (int mi = 0; mi < 2; ++mi)
#pragma unroll
    for (int ni = 0; ni < 4; ++ni)
      acc[mi][ni] = (v8f){0.f, 0.f, 0.f, 0.f, 0.f, 0.f, 0.f, 0.f};

  // ---- 16 WMMAs against one staged 128x64 LDS tile pair ----
  auto compute_tile = [&](const unsigned short* Asb, const unsigned short* Bsb) {
#pragma unroll
    for (int ks = 0; ks < 2; ++ks) {
      FragBF a[2], b[4];
#pragma unroll
      for (int mi = 0; mi < 2; ++mi) {
        // A 16x32: lanes 0-15 M=lr K={0..7,16..23}; lanes 16-31 K={8..15,24..31}
        const u32x4* pa = (const u32x4*)&Asb[(mw + mi * 16 + lr) * LDT];
        a[mi].u[0] = pa[4 * ks + lh];
        a[mi].u[1] = pa[4 * ks + 2 + lh];
      }
#pragma unroll
      for (int ni = 0; ni < 4; ++ni) {
        // B 32x16 from row-major NT operand: lane n=lr, K={0..7,16..23} (lh=0) / {8..15,24..31} (lh=1)
        const u32x4* pb = (const u32x4*)&Bsb[(nw + ni * 16 + lr) * LDT];
        b[ni].u[0] = pb[4 * ks + lh];
        b[ni].u[1] = pb[4 * ks + 2 + lh];
      }
#pragma unroll
      for (int mi = 0; mi < 2; ++mi)
#pragma unroll
        for (int ni = 0; ni < 4; ++ni)
          acc[mi][ni] = __builtin_amdgcn_wmma_f32_16x16x32_bf16(
              false, a[mi].v, false, b[ni].v, (short)0, acc[mi][ni], false, false);
    }
  };

  if constexpr (MODE <= 1) {
    // ---- f32 inputs: VGPR software pipeline, single LDS buffer ----
    const float* A = (const float*)Aptr;
    const float* B = (const float*)Bptr;
    unsigned short* As = sm;
    unsigned short* Bs = sm + TILE_E;

    f32x4 ra[4][2], rb[4][2];
    auto load_regs = [&](int kk) {
#pragma unroll
      for (int j = 0; j < 4; ++j) {
        const int id = tid + j * 256;
        const int r2 = id >> 3;        // 0..127
        const int q2 = id & 7;         // 8-elem chunk
        const float* pa = A + (m0 + r2) * (size_t)Kdim + kk + q2 * 8;
        const float* pb = B + (n0 + r2) * (size_t)Kdim + kk + q2 * 8;
        ra[j][0] = *(const f32x4*)pa;  ra[j][1] = *(const f32x4*)(pa + 4);
        rb[j][0] = *(const f32x4*)pb;  rb[j][1] = *(const f32x4*)(pb + 4);
        if (kk + BKT < Kdim) {
          __builtin_prefetch(pa + BKT, 0, 3);   // -> global_prefetch_b8
          __builtin_prefetch(pb + BKT, 0, 3);
        }
      }
    };

    load_regs(0);
    for (int kk = 0; kk < Kdim; kk += BKT) {
      __syncthreads();   // previous compute done reading LDS
#pragma unroll
      for (int j = 0; j < 4; ++j) {
        const int id = tid + j * 256;
        const int r2 = id >> 3;
        const int q2 = id & 7;
        u32x4 ap, bp;
#pragma unroll
        for (int h = 0; h < 2; ++h) {
          ap[2 * h + 0] = (unsigned)f32_to_bf16(ra[j][h][0]) |
                          ((unsigned)f32_to_bf16(ra[j][h][1]) << 16);
          ap[2 * h + 1] = (unsigned)f32_to_bf16(ra[j][h][2]) |
                          ((unsigned)f32_to_bf16(ra[j][h][3]) << 16);
          bp[2 * h + 0] = (unsigned)f32_to_bf16(rb[j][h][0]) |
                          ((unsigned)f32_to_bf16(rb[j][h][1]) << 16);
          bp[2 * h + 1] = (unsigned)f32_to_bf16(rb[j][h][2]) |
                          ((unsigned)f32_to_bf16(rb[j][h][3]) << 16);
        }
        *(u32x4*)&As[r2 * LDT + q2 * 8] = ap;
        *(u32x4*)&Bs[r2 * LDT + q2 * 8] = bp;
      }
      if (kk + BKT < Kdim) load_regs(kk + BKT);   // overlap with compute
      __syncthreads();
      compute_tile(As, Bs);
    }
  } else {
    // ---- bf16 inputs: async global->LDS, double-buffered ----
    const unsigned short* A = (const unsigned short*)Aptr;
    const unsigned short* B = (const unsigned short*)Bptr;

    auto issue = [&](int kk, int buf) {
      unsigned short* Asb = sm + buf * (2 * TILE_E);
      unsigned short* Bsb = Asb + TILE_E;
#pragma unroll
      for (int j = 0; j < 4; ++j) {
        const int id = tid + j * 256;
        const int r2 = id >> 3;
        const int q2 = id & 7;
        async_copy_b128((unsigned)(size_t)&Asb[r2 * LDT + q2 * 8],
                        A + (m0 + r2) * (size_t)Kdim + kk + q2 * 8);
        async_copy_b128((unsigned)(size_t)&Bsb[r2 * LDT + q2 * 8],
                        B + (n0 + r2) * (size_t)Kdim + kk + q2 * 8);
      }
    };

    const int T = Kdim / BKT;
    issue(0, 0);
    for (int t = 0; t < T; ++t) {
      const int cur = t & 1;
      if (t + 1 < T) {
        issue((t + 1) * BKT, cur ^ 1);  // prefetch next tile into other buffer
        wait_asynccnt<8>();             // in-order: oldest 8 (tile t) retired
      } else {
        wait_asynccnt<0>();
      }
      __syncthreads();
      const unsigned short* Asb = sm + cur * (2 * TILE_E);
      compute_tile(Asb, Asb + TILE_E);
      __syncthreads();
    }
  }

  // ---- epilogue ----
  // C/D layout: VGPR i, lane -> (m = i + 8*lh, n = lr) within each 16x16 tile.
  if constexpr (MODE == MODE_PROJ || MODE == MODE_SCORE) {
    // stage bf16 tile in LDS, then coalesced b128 stores
    unsigned short* st = sm;
    __syncthreads();
#pragma unroll
    for (int ni = 0; ni < 4; ++ni) {
      const int n = nw + ni * 16 + lr;
      float bv = 0.f;
      if constexpr (MODE == MODE_PROJ) bv = bias[n0 + n];
#pragma unroll
      for (int mi = 0; mi < 2; ++mi)
#pragma unroll
        for (int i = 0; i < 8; ++i) {
          const int m = mw + mi * 16 + i + 8 * lh;
          float v = acc[mi][ni][i];
          if constexpr (MODE == MODE_PROJ) v += bv; else v *= scale;
          st[m * SP16 + n] = f32_to_bf16(v);
        }
    }
    __syncthreads();
    unsigned short* O = (unsigned short*)Optr;
#pragma unroll
    for (int j = 0; j < 8; ++j) {
      const int id = tid + j * 256;
      const int r = id >> 4;           // 0..127
      const int q = id & 15;           // 16 chunks of 8 bf16 per row
      *(u32x4*)&O[(m0 + r) * (size_t)ldo + n0 + q * 8] =
          *(const u32x4*)&st[r * SP16 + q * 8];
    }
  } else if constexpr (MODE == MODE_PROJ_T) {
    unsigned short* O = (unsigned short*)Optr;
#pragma unroll
    for (int ni = 0; ni < 4; ++ni) {
      const size_t n = n0 + nw + ni * 16 + lr;
      const float bv = bias[n];
#pragma unroll
      for (int mi = 0; mi < 2; ++mi) {
        u32x4 p;
#pragma unroll
        for (int j = 0; j < 4; ++j) {
          p[j] = (unsigned)f32_to_bf16(acc[mi][ni][2 * j] + bv) |
                 ((unsigned)f32_to_bf16(acc[mi][ni][2 * j + 1] + bv) << 16);
        }
        const size_t mbase = m0 + mw + mi * 16 + 8 * lh;  // 8 consecutive m
        *(u32x4*)&O[n * (size_t)ldo + mbase] = p;          // one b128 store
      }
    }
  } else {  // MODE_OUT: stage f32 tile in LDS, then coalesced b128 stores
    float* stf = (float*)sm;
    __syncthreads();
#pragma unroll
    for (int mi = 0; mi < 2; ++mi)
#pragma unroll
      for (int i = 0; i < 8; ++i) {
        const int m = mw + mi * 16 + i + 8 * lh;
        const float inv = 1.0f / rowsum[m0 + m];
#pragma unroll
        for (int ni = 0; ni < 4; ++ni)
          stf[m * SP32 + nw + ni * 16 + lr] = acc[mi][ni][i] * inv;
      }
    __syncthreads();
    float* O = (float*)Optr;
#pragma unroll
    for (int j = 0; j < 16; ++j) {
      const int id = tid + j * 256;
      const int r = id >> 5;           // 0..127
      const int q = id & 31;           // 32 chunks of 4 f32 per row
      *(f32x4*)&O[(m0 + r) * (size_t)ldo + n0 + q * 4] =
          *(const f32x4*)&stf[r * SP32 + q * 4];
    }
  }
}

// One workgroup per row: the whole 8192-element bf16 row lives in registers
// (32 elems/thread). max -> exp -> sum -> store, no atomics, S read once.
__global__ __launch_bounds__(256) void softmax_rows(
    unsigned short* __restrict__ S, float* __restrict__ rowsum, int N)
{
  __shared__ float red[256];
  const int tid = threadIdx.x;
  unsigned short* row = S + (size_t)blockIdx.x * (size_t)N;

  u32x4 d[4];
  const u32x4* src = (const u32x4*)(row + tid * 32);
#pragma unroll
  for (int c = 0; c < 4; ++c) d[c] = src[c];

  float v[32];
  float mx = -3.402823466e38f;
#pragma unroll
  for (int c = 0; c < 4; ++c) {
#pragma unroll
    for (int j = 0; j < 4; ++j) {
      const unsigned int u = d[c][j];
      const float lo = bf16_to_f32(u & 0xFFFFu);
      const float hi = bf16_to_f32(u >> 16);
      v[c * 8 + 2 * j]     = lo;
      v[c * 8 + 2 * j + 1] = hi;
      mx = fmaxf(mx, fmaxf(lo, hi));
    }
  }

  red[tid] = mx;
  __syncthreads();
  for (int off = 128; off > 0; off >>= 1) {
    if (tid < off) red[tid] = fmaxf(red[tid], red[tid + off]);
    __syncthreads();
  }
  const float rmax = red[0];
  __syncthreads();

  float s = 0.f;
#pragma unroll
  for (int i = 0; i < 32; ++i) {
    const float e = __expf(v[i] - rmax);   // v_exp_f32
    v[i] = e;
    s += e;
  }

  red[tid] = s;
  __syncthreads();
  for (int off = 128; off > 0; off >>= 1) {
    if (tid < off) red[tid] += red[tid + off];
    __syncthreads();
  }
  if (tid == 0) rowsum[blockIdx.x] = red[0];

  u32x4 o[4];
#pragma unroll
  for (int c = 0; c < 4; ++c) {
#pragma unroll
    for (int j = 0; j < 4; ++j) {
      o[c][j] = (unsigned)f32_to_bf16(v[c * 8 + 2 * j]) |
                ((unsigned)f32_to_bf16(v[c * 8 + 2 * j + 1]) << 16);
    }
  }
  u32x4* dst = (u32x4*)(row + tid * 32);
#pragma unroll
  for (int c = 0; c < 4; ++c) dst[c] = o[c];
}

extern "C" void kernel_launch(void* const* d_in, const int* in_sizes, int n_in,
                              void* d_out, int out_size, void* d_ws, size_t ws_size,
                              hipStream_t stream) {
  (void)in_sizes; (void)n_in; (void)out_size; (void)ws_size;
  const float* m1 = (const float*)d_in[0];   // [8192,1024]
  const float* m2 = (const float*)d_in[1];   // [8192,1024]
  const float* Wq = (const float*)d_in[2];   // [1024,1024]
  const float* bq = (const float*)d_in[3];
  const float* Wk = (const float*)d_in[4];
  const float* bk = (const float*)d_in[5];
  const float* Wv = (const float*)d_in[6];
  const float* bv = (const float*)d_in[7];

  char* ws = (char*)d_ws;
  unsigned short* Qb = (unsigned short*)(ws);                      // 16 MB
  unsigned short* Kb = (unsigned short*)(ws + (16ull << 20));      // 16 MB
  unsigned short* VT = (unsigned short*)(ws + (32ull << 20));      // 16 MB
  unsigned short* S  = (unsigned short*)(ws + (48ull << 20));      // 128 MB (L2-resident)
  float* rowsum      = (float*)(ws + (176ull << 20));              // 32 KB

  const dim3 blk(256);
  const float scale = 0.03125f;  // 1/sqrt(1024)

  // Q = bf16(m1 @ Wq^T + bq)             [8192 x 1024]
  gemm_nt<MODE_PROJ><<<dim3(8, 64), blk, 0, stream>>>(
      m1, Wq, bq, nullptr, Qb, 1024, 1024, 1.0f);
  // K = bf16(m2 @ Wk^T + bk)             [8192 x 1024]
  gemm_nt<MODE_PROJ><<<dim3(8, 64), blk, 0, stream>>>(
      m2, Wk, bk, nullptr, Kb, 1024, 1024, 1.0f);
  // VT = bf16((m2 @ Wv^T + bv))^T        [1024 x 8192]
  gemm_nt<MODE_PROJ_T><<<dim3(8, 64), blk, 0, stream>>>(
      m2, Wv, bv, nullptr, VT, 1024, 8192, 1.0f);
  // S = bf16(scale * Q @ K^T)            [8192 x 8192], stays hot in 192MB L2
  gemm_nt<MODE_SCORE><<<dim3(64, 64), blk, 0, stream>>>(
      Qb, Kb, nullptr, nullptr, S, 1024, 8192, scale);
  // P = exp(S - rowmax) in place; rowsum[m] = sum_j P[m,j]
  softmax_rows<<<dim3(8192), blk, 0, stream>>>(S, rowsum, 8192);
  // out = (P @ VT^T) / rowsum            [8192 x 1024] f32
  gemm_nt<MODE_OUT><<<dim3(8, 64), blk, 0, stream>>>(
      S, VT, nullptr, rowsum, d_out, 8192, 1024, 1.0f);
}
